// LSTMCell_87969520156813
// MI455X (gfx1250) — matrix operand order repaired and checked by reading, not compiled, not run on hardware
//
#include <hip/hip_runtime.h>

// ---- ext-vector types ----
typedef __attribute__((ext_vector_type(16))) __bf16 v16bf;
typedef __attribute__((ext_vector_type(8)))  float  v8f;
typedef __attribute__((ext_vector_type(4)))  float  v4f;
typedef __attribute__((ext_vector_type(4)))  int    v4i;
typedef __attribute__((ext_vector_type(2)))  int    v2i;

#define B_SZ   4096
#define D_IN   2048
#define D_HID  2048
#define K_TOT  (D_IN + D_HID)     // 4096
#define KT     32                 // K depth per WMMA step (bf16)
#define NKT    (K_TOT / KT)       // 128 K-steps
#define MT     128                // block tile: batch rows
#define HT     64                 // block tile: h columns (per gate; x4 gates)
#define BROWS  (4 * HT)           // 256 weight rows staged per step
#define PITCH  40                 // LDS row pitch in ushorts (80B; conflict-free frag loads)

union Frag { v4i q[2]; v16bf v; };

// pack two fp32 -> one dword of two bf16 (truncate) with a single v_perm_b32
__device__ __forceinline__ unsigned int pk2(float hi, float lo) {
  return __builtin_amdgcn_perm(__float_as_uint(hi), __float_as_uint(lo), 0x07060302u);
}

__device__ __forceinline__ float fast_sigmoid(float x) {
  return 1.0f / (1.0f + __expf(-x));
}
__device__ __forceinline__ float fast_tanh(float x) {
  return 2.0f / (1.0f + __expf(-2.0f * x)) - 1.0f;
}

__global__ __launch_bounds__(256)
void lstm_cell_wmma(const float* __restrict__ x,      const float* __restrict__ mask,
                    const float* __restrict__ last_c, const float* __restrict__ last_h,
                    const float* __restrict__ W0, const float* __restrict__ W1,
                    const float* __restrict__ W2, const float* __restrict__ W3,
                    const float* __restrict__ U0, const float* __restrict__ U1,
                    const float* __restrict__ U2, const float* __restrict__ U3,
                    const float* __restrict__ bi, const float* __restrict__ bfp,
                    const float* __restrict__ bo, const float* __restrict__ bc,
                    float* __restrict__ out)
{
  // A tile: 128 batch rows x 32 K (bf16). B tile: 256 weight rows (4 gates x 64 h) x 32 K.
  __shared__ __align__(16) unsigned short As[2][MT][PITCH];
  __shared__ __align__(16) unsigned short Bs[2][BROWS][PITCH];

  const int t    = threadIdx.x;
  const int lane = t & 31;
  const int wave = t >> 5;        // 0..7
  const int wm   = wave & 1;      // M group: 64 rows each
  const int wh   = wave >> 1;     // H group (0..3): 16 cols each
  const int lrow = lane & 15;
  const int lhi  = lane >> 4;

  const int bm = blockIdx.x;      // 0..31 (M tiles)
  const int bh = blockIdx.y;      // 0..31 (H tiles)

  // tile staging: 8 lanes per row, one float4 each -> fully coalesced 128B rows
  const int lr8 = t >> 3;         // 0..31: base row within each 32-row group
  const int kc  = (t & 7) * 4;    // K offset (elements) within the 32-wide tile

  v4f ar[4];   // A rows: 32*j + lr8, j=0..3
  v4f br[8];   // B rows: 32*j + lr8, j=0..7 (gate = j>>1, constant-folded)

  auto load_tiles = [&](int kt) {
    const int kk = kt * KT;
    const float *abase, *g0, *g1, *g2, *g3;
    int ko;
    if (kk < D_IN) { abase = x;      g0 = W0; g1 = W1; g2 = W2; g3 = W3; ko = kk; }
    else           { abase = last_h; g0 = U0; g1 = U1; g2 = U2; g3 = U3; ko = kk - D_IN; }
#pragma unroll
    for (int j = 0; j < 4; ++j) {
      const int row = bm * MT + 32 * j + lr8;
      ar[j] = *(const v4f*)(abase + (size_t)row * D_IN + ko + kc);
    }
#pragma unroll
    for (int j = 0; j < 8; ++j) {
      const float* wb = (j < 2) ? g0 : (j < 4) ? g1 : (j < 6) ? g2 : g3;
      const int hrow  = bh * HT + 32 * (j & 1) + lr8;
      br[j] = *(const v4f*)(wb + (size_t)hrow * D_IN + ko + kc);
    }
  };

  auto store_tiles = [&](int buf) {
#pragma unroll
    for (int j = 0; j < 4; ++j) {
      v2i a = { (int)pk2(ar[j][1], ar[j][0]), (int)pk2(ar[j][3], ar[j][2]) };
      *(v2i*)&As[buf][32 * j + lr8][kc] = a;
    }
#pragma unroll
    for (int j = 0; j < 8; ++j) {
      v2i b = { (int)pk2(br[j][1], br[j][0]), (int)pk2(br[j][3], br[j][2]) };
      *(v2i*)&Bs[buf][32 * j + lr8][kc] = b;
    }
  };

  // bias folded into accumulator init: C element (M,N) needs bias[N] only,
  // and N == lane%16 is constant across all 8 C VGPRs.
  const int   colh = bh * HT + wh * 16 + lrow;   // h index 0..2047
  const float bias4[4] = { bi[colh], bfp[colh], bo[colh], bc[colh] };

  v8f acc[4][4];   // [m_subtile][gate]
#pragma unroll
  for (int ms = 0; ms < 4; ++ms)
#pragma unroll
    for (int g = 0; g < 4; ++g)
#pragma unroll
      for (int r = 0; r < 8; ++r)
        acc[ms][g][r] = bias4[g];

  load_tiles(0);
  store_tiles(0);
  __syncthreads();

  int cur = 0;
  for (int kt = 0; kt < NKT; ++kt) {
    if (kt + 1 < NKT) load_tiles(kt + 1);   // issue next global loads early

    // B fragments: lane<16 holds K[0..7],[16..23]; lane>=16 K[8..15],[24..31] of column n
    Frag bfr[4];
#pragma unroll
    for (int g = 0; g < 4; ++g) {
      const unsigned short* p = &Bs[cur][g * HT + wh * 16 + lrow][lhi * 8];
      bfr[g].q[0] = *(const v4i*)p;
      bfr[g].q[1] = *(const v4i*)(p + 16);
    }

    // A fragments in two halves to cap register pressure:
    // ISA 16-bit A layout — lane<16 holds K[0..7],[16..23]; lane>=16 K[8..15],[24..31]
#pragma unroll
    for (int half = 0; half < 2; ++half) {
      Frag af0, af1;
      const unsigned short* p0 = &As[cur][wm * 64 + (half * 2 + 0) * 16 + lrow][lhi * 8];
      const unsigned short* p1 = &As[cur][wm * 64 + (half * 2 + 1) * 16 + lrow][lhi * 8];
      af0.q[0] = *(const v4i*)p0;  af0.q[1] = *(const v4i*)(p0 + 16);
      af1.q[0] = *(const v4i*)p1;  af1.q[1] = *(const v4i*)(p1 + 16);
#pragma unroll
      for (int g = 0; g < 4; ++g) {
        acc[half * 2 + 0][g] = __builtin_amdgcn_wmma_f32_16x16x32_bf16(
            false, af0.v, false, bfr[g].v, (short)0, acc[half * 2 + 0][g], false, false);
        acc[half * 2 + 1][g] = __builtin_amdgcn_wmma_f32_16x16x32_bf16(
            false, af1.v, false, bfr[g].v, (short)0, acc[half * 2 + 1][g], false, false);
      }
    }

    if (kt + 1 < NKT) store_tiles(cur ^ 1);  // fill other buffer while this one was read
    __syncthreads();
    cur ^= 1;
  }

  // ---- fused LSTM epilogue ----
  // C/D layout: VGPR r, lane<16 -> row r, col lane; lane>=16 -> row 8+r, col lane-16
#pragma unroll
  for (int ms = 0; ms < 4; ++ms) {
    const int row0 = bm * MT + wm * 64 + ms * 16 + lhi * 8;
#pragma unroll
    for (int r = 0; r < 8; ++r) {
      const int    row = row0 + r;
      const size_t idx = (size_t)row * D_HID + colh;
      const float gi = acc[ms][0][r];
      const float gf = acc[ms][1][r];
      const float go = acc[ms][2][r];
      const float gc = acc[ms][3][r];
      const float iv = fast_sigmoid(gi);
      const float fv = fast_sigmoid(gf);
      const float ov = fast_sigmoid(go);
      const float ch = fast_tanh(gc);
      const float lc = last_c[idx];
      float c  = iv * ch + fv * lc;
      float h  = ov * fast_tanh(c);
      const float mk = mask[row];
      const float lh = last_h[idx];
      c = lc + mk * (c - lc);
      h = lh + mk * (h - lh);
      out[idx] = c;
      out[(size_t)B_SZ * D_HID + idx] = h;
    }
  }
}

extern "C" void kernel_launch(void* const* d_in, const int* in_sizes, int n_in,
                              void* d_out, int out_size, void* d_ws, size_t ws_size,
                              hipStream_t stream) {
  (void)in_sizes; (void)n_in; (void)out_size; (void)d_ws; (void)ws_size;
  const float* x      = (const float*)d_in[0];
  const float* mask   = (const float*)d_in[1];
  const float* last_c = (const float*)d_in[2];
  const float* last_h = (const float*)d_in[3];
  const float* Wi     = (const float*)d_in[4];
  const float* Wf     = (const float*)d_in[5];
  const float* Wo     = (const float*)d_in[6];
  const float* Wc     = (const float*)d_in[7];
  const float* Ui     = (const float*)d_in[8];
  const float* Uf     = (const float*)d_in[9];
  const float* Uo     = (const float*)d_in[10];
  const float* Uc     = (const float*)d_in[11];
  const float* bi     = (const float*)d_in[12];
  const float* bf     = (const float*)d_in[13];
  const float* bo     = (const float*)d_in[14];
  const float* bc     = (const float*)d_in[15];

  dim3 grid(B_SZ / MT, D_HID / HT);   // (32, 32): x-major shares the weight tile in L2
  dim3 block(256);                    // 8 wave32 waves
  lstm_cell_wmma<<<grid, block, 0, stream>>>(x, mask, last_c, last_h,
                                             Wi, Wf, Wo, Wc, Ui, Uf, Uo, Uc,
                                             bi, bf, bo, bc, (float*)d_out);
}
